// PPM_6743098655522
// MI455X (gfx1250) — matrix operand order repaired and checked by reading, not compiled, not run on hardware
//
#include <hip/hip_runtime.h>
#include <hip/hip_bf16.h>

typedef _Float16 half_t;
typedef __attribute__((ext_vector_type(16))) _Float16 v16h;
typedef __attribute__((ext_vector_type(8)))  float    v8f;

#define BATCH 4
#define CCH   256            // channels
#define NN    4096           // D*W*H
#define NCHUNK 128           // n-rows per block in fused kernel

union U16h {
    v16h v;
    uint4 q[2];
};
union U8h {
    uint4 q;
    half_t h[8];
};
union U8f {
    v8f f;
    uint4 q[2];
};

// ---------------------------------------------------------------------------
// Kernel 0: convert W (f32, [C][C]) -> Wh (f16, [C][C], row-major over k)
// ---------------------------------------------------------------------------
__global__ void PPM_convert_w(const float* __restrict__ W, half_t* __restrict__ Wh) {
    int i = blockIdx.x * blockDim.x + threadIdx.x;   // 0 .. 65535
    Wh[i] = (half_t)W[i];
}

// ---------------------------------------------------------------------------
// Kernel 1: per (b, n): column norm over c; write
//   xT [b][n][c] = (f16) x[b][c][n]          (raw transpose, feeds t-GEMM B)
//   xnT[b][n][c] = (f16)(x[b][c][n] / max(||x[:,n]||, eps))   (feeds sim)
// ---------------------------------------------------------------------------
__global__ __launch_bounds__(256) void PPM_prep(const float* __restrict__ x,
                                                half_t* __restrict__ xT,
                                                half_t* __restrict__ xnT) {
    int b = blockIdx.y;
    int n = blockIdx.x * 256 + threadIdx.x;
    const float* xb = x + (size_t)b * CCH * NN + n;   // stride NN over c

    float ss = 0.f;
    #pragma unroll 8
    for (int c = 0; c < CCH; ++c) {
        float v = xb[(size_t)c * NN];
        ss += v * v;
    }
    float rn = 1.f / fmaxf(__builtin_sqrtf(ss), 1e-8f);

    half_t* trow = xT  + ((size_t)b * NN + n) * CCH;
    half_t* nrow = xnT + ((size_t)b * NN + n) * CCH;
    for (int c0 = 0; c0 < CCH; c0 += 8) {
        U8h raw, nrm;
        #pragma unroll
        for (int j = 0; j < 8; ++j) {
            float v = xb[(size_t)(c0 + j) * NN];
            raw.h[j] = (half_t)v;
            nrm.h[j] = (half_t)(v * rn);
        }
        *(uint4*)(trow + c0) = raw.q;
        *(uint4*)(nrow + c0) = nrm.q;
    }
}

// ---------------------------------------------------------------------------
// Kernel 2: t = W * xf + bias  via WMMA.
//   t[c,m] = sum_k W[c,k] x[b,k,m] + bias[c]
//   A = Wh rows (c), K=k contiguous.  B = xT rows (m), K=k contiguous.
//   Output stored as tB[b][c][m] (f16) so the fused kernel's B fetch is
//   contiguous in m.
// Each wave computes one 16x16 (c,m) tile; 8 waves / block.
// ---------------------------------------------------------------------------
__global__ __launch_bounds__(256) void PPM_tgemm(const half_t* __restrict__ Wh,
                                                 const half_t* __restrict__ xT,
                                                 const float* __restrict__ bias,
                                                 half_t* __restrict__ tB) {
    int b    = blockIdx.y;
    int wave = threadIdx.x >> 5;
    int lane = threadIdx.x & 31;
    int lhi  = lane >> 4;       // 0 / 1
    int llo  = lane & 15;

    int tid = blockIdx.x * 8 + wave;          // 0 .. 4095 tiles per batch
    int c0  = (tid & 15) * 16;
    int m0  = (tid >> 4) * 16;

    const half_t* xb = xT + (size_t)b * NN * CCH;

    v8f acc;
    #pragma unroll
    for (int i = 0; i < 8; ++i) acc[i] = 0.f;

    #pragma unroll
    for (int kc = 0; kc < 8; ++kc) {
        // A: W rows c0+llo, K-base = lhi*16 within this 32-wide K chunk
        const half_t* ap = Wh + (size_t)(c0 + llo) * CCH + kc * 32 + lhi * 16;
        U16h A;
        A.q[0] = *(const uint4*)ap;          // K base .. base+7
        A.q[1] = *(const uint4*)(ap + 8);    // K base+8 .. base+15
        // B: xT rows m0+llo, K-base = lhi*16
        const half_t* bp = xb + (size_t)(m0 + llo) * CCH + kc * 32 + lhi * 16;
        U16h Bm;
        Bm.q[0] = *(const uint4*)bp;
        Bm.q[1] = *(const uint4*)(bp + 8);
        acc = __builtin_amdgcn_wmma_f32_16x16x32_f16(false, A.v, false, Bm.v,
                                                     (short)0, acc, false, false);
    }

    half_t* tb = tB + (size_t)b * CCH * NN;
    #pragma unroll
    for (int r = 0; r < 8; ++r) {
        int c = c0 + r + 8 * lhi;            // D row
        int m = m0 + llo;                    // D col
        tb[(size_t)c * NN + m] = (half_t)(acc[r] + bias[c]);
    }
}

// ---------------------------------------------------------------------------
// Kernel 3 (fused): out[b,c,n] = sum_m relu(xn[:,n].xn[:,m])^2 * t[c,m]
// Block = (batch, 128 n-rows). Wave w owns n-subtile rows [n0+16w, +16) and
// the full 16x256 output strip (16 f32 accum tiles).
// Per 32-m chunk: 16 sim WMMAs -> relu^2 -> f16 via LDS lane-transpose ->
// 16 output WMMAs.
// ---------------------------------------------------------------------------
__global__ __launch_bounds__(256) void PPM_fused(const half_t* __restrict__ xnT,
                                                 const half_t* __restrict__ tB,
                                                 float* __restrict__ out) {
    __shared__ alignas(32) half_t sbuf[8][16][32];   // per-wave S tile (f16)

    int b    = blockIdx.y;
    int n0   = blockIdx.x * NCHUNK;
    int wave = threadIdx.x >> 5;
    int lane = threadIdx.x & 31;
    int lhi  = lane >> 4;
    int llo  = lane & 15;
    int nrow0 = n0 + wave * 16;

    const half_t* xb = xnT + (size_t)b * NN * CCH;
    const half_t* tb = tB  + (size_t)b * CCH * NN;

    // Cache Q (A operand for sim) for this wave's 16 n-rows: 8 K-chunks.
    v16h Q[8];
    #pragma unroll
    for (int kc = 0; kc < 8; ++kc) {
        const half_t* rp = xb + (size_t)(nrow0 + llo) * CCH + kc * 32 + lhi * 16;
        U16h u;
        u.q[0] = *(const uint4*)rp;
        u.q[1] = *(const uint4*)(rp + 8);
        Q[kc] = u.v;
    }

    // Output accumulators: 16 c-tiles of 16x16 f32.
    v8f acc[16];
    #pragma unroll
    for (int t = 0; t < 16; ++t)
        #pragma unroll
        for (int i = 0; i < 8; ++i) acc[t][i] = 0.f;

    for (int mc = 0; mc < NN; mc += 32) {
        // Prefetch next chunk's operand lines into near cache while the
        // 32 WMMAs of this iteration execute (global_prefetch_b8).
        if (mc + 32 < NN) {
            __builtin_prefetch(xb + (size_t)(mc + 32 + llo) * CCH + lhi * 16, 0, 1);
            __builtin_prefetch(xb + (size_t)(mc + 48 + llo) * CCH + lhi * 16, 0, 1);
            __builtin_prefetch(tb + (size_t)(wave * 16 + llo) * NN + mc + 32 + lhi * 16, 0, 1);
        }

        // ---- sim tiles S0 (m = mc..mc+15), S1 (m = mc+16..mc+31) ----
        v8f s0, s1;
        #pragma unroll
        for (int i = 0; i < 8; ++i) { s0[i] = 0.f; s1[i] = 0.f; }

        #pragma unroll
        for (int kc = 0; kc < 8; ++kc) {
            const half_t* bp0 = xb + (size_t)(mc + llo) * CCH + kc * 32 + lhi * 16;
            U16h B0;
            B0.q[0] = *(const uint4*)bp0;
            B0.q[1] = *(const uint4*)(bp0 + 8);
            s0 = __builtin_amdgcn_wmma_f32_16x16x32_f16(false, Q[kc], false, B0.v,
                                                        (short)0, s0, false, false);
            const half_t* bp1 = xb + (size_t)(mc + 16 + llo) * CCH + kc * 32 + lhi * 16;
            U16h B1;
            B1.q[0] = *(const uint4*)bp1;
            B1.q[1] = *(const uint4*)(bp1 + 8);
            s1 = __builtin_amdgcn_wmma_f32_16x16x32_f16(false, Q[kc], false, B1.v,
                                                        (short)0, s1, false, false);
        }

        // ---- relu()^2, cvt f16, stage through LDS to reach A-operand layout
        #pragma unroll
        for (int r = 0; r < 8; ++r) {
            float p0 = fmaxf(s0[r], 0.f); p0 *= p0;
            float p1 = fmaxf(s1[r], 0.f); p1 *= p1;
            sbuf[wave][r + 8 * lhi][llo]      = (half_t)p0;   // cols mc..+16
            sbuf[wave][r + 8 * lhi][16 + llo] = (half_t)p1;   // cols mc+16..+32
        }
        // intra-wave LDS RAW: hw keeps DS in order; fence the compiler + cnt
        asm volatile("s_wait_dscnt 0" ::: "memory");

        // A operand = S (16 n-rows x 32 m-cols, f16)
        const half_t* sp = &sbuf[wave][llo][lhi * 16];
        U16h Am;
        Am.q[0] = *(const uint4*)sp;          // K base .. +7
        Am.q[1] = *(const uint4*)(sp + 8);    // K base+8 .. +15
        v16h Amat = Am.v;

        // ---- out tiles: acc[t] += S * tB[c-tile t][mc..mc+32] ----
        #pragma unroll
        for (int t = 0; t < 16; ++t) {
            const half_t* vp = tb + (size_t)(t * 16 + llo) * NN + mc + lhi * 16;
            U16h Bv;
            Bv.q[0] = *(const uint4*)vp;
            Bv.q[1] = *(const uint4*)(vp + 8);
            acc[t] = __builtin_amdgcn_wmma_f32_16x16x32_f16(false, Amat, false, Bv.v,
                                                            (short)0, acc[t], false, false);
        }
    }

    // ---- store: out[b][c][n].  In the D layout a lane's 8 accum registers
    // are CONSECUTIVE n values (M = r + 8*lhi), so each tile is exactly two
    // 16-byte contiguous stores per lane: global_store_b128.
    float* ob = out + (size_t)b * CCH * NN;
    #pragma unroll
    for (int t = 0; t < 16; ++t) {
        int c = t * 16 + llo;
        float* op = ob + (size_t)c * NN + nrow0 + 8 * lhi;
        U8f u;
        u.f = acc[t];
        *(uint4*)(op)     = u.q[0];   // n-local 8*lhi .. +3
        *(uint4*)(op + 4) = u.q[1];   // n-local 8*lhi+4 .. +7
    }
}

// ---------------------------------------------------------------------------
extern "C" void kernel_launch(void* const* d_in, const int* in_sizes, int n_in,
                              void* d_out, int out_size, void* d_ws, size_t ws_size,
                              hipStream_t stream) {
    const float* x    = (const float*)d_in[0];   // [4][256][4096] f32
    const float* W    = (const float*)d_in[1];   // [256][256] f32
    const float* bias = (const float*)d_in[2];   // [256] f32
    float* out        = (float*)d_out;           // [4][256][4096] f32

    char* ws = (char*)d_ws;
    half_t* xT  = (half_t*)ws;                            // 8 MB: raw transpose
    half_t* xnT = (half_t*)(ws + ((size_t)8  << 20));     // 8 MB: normalized transpose
    half_t* tB  = (half_t*)(ws + ((size_t)16 << 20));     // 8 MB: t in [b][c][m]
    half_t* Wh  = (half_t*)(ws + ((size_t)24 << 20));     // 128 KB: W in f16

    PPM_convert_w<<<dim3(CCH * CCH / 256), 256, 0, stream>>>(W, Wh);
    PPM_prep     <<<dim3(NN / 256, BATCH), 256, 0, stream>>>(x, xT, xnT);
    PPM_tgemm    <<<dim3((CCH / 16) * (NN / 16) / 8, BATCH), 256, 0, stream>>>(Wh, xT, bias, tB);
    PPM_fused    <<<dim3(NN / NCHUNK, BATCH), 256, 0, stream>>>(xnT, tB, out);
}
